// CausalSelfAttention_26568667693689
// MI455X (gfx1250) — matrix operand order repaired and checked by reading, not compiled, not run on hardware
//
#include <hip/hip_runtime.h>
#include <hip/hip_bf16.h>

// ---------------------------------------------------------------------------
// CDNA5 (gfx1250) causal self-attention, bf16 WMMA (f32 accumulate), wave32.
// GEMM K-loops use explicit ping-pong double buffering (no register copies).
// ---------------------------------------------------------------------------

typedef __attribute__((ext_vector_type(16))) __bf16 v16bf;
typedef __attribute__((ext_vector_type(8)))  float  v8f;
typedef __attribute__((ext_vector_type(4)))  unsigned int v4u;

#define WMMA_BF16(a, b, c) \
    __builtin_amdgcn_wmma_f32_16x16x32_bf16(false, (a), false, (b), (short)0, (c), false, false)

// Problem constants
static constexpr int BB   = 2;
static constexpr int TT   = 2048;
static constexpr int CC   = 768;
static constexpr int HH   = 12;
static constexpr int DD   = 64;
static constexpr int MM   = BB * TT;        // 4096 rows
static constexpr int N_QKV = 3 * CC;        // 2304
static constexpr float ATTN_SCALE = 0.70710678118654752f; // 1/sqrt(B=2), per reference

__device__ __forceinline__ unsigned short f2bf(float f) {
    unsigned int u = __float_as_uint(f);
    u += 0x7FFFu + ((u >> 16) & 1u);        // round to nearest even
    return (unsigned short)(u >> 16);
}

// Load a 16-element bf16 fragment as two 16-byte chunks (both 16B aligned).
__device__ __forceinline__ v16bf load_frag(const unsigned short* p0, const unsigned short* p1) {
    union { v4u q[2]; v16bf v; } u;
    u.q[0] = *(const v4u*)p0;
    u.q[1] = *(const v4u*)p1;
    return u.v;
}

// ---------------------------------------------------------------------------
// Stage 0: fp32 -> bf16 conversion
// ---------------------------------------------------------------------------
__global__ void __launch_bounds__(256) cvt_f32_bf16(const float* __restrict__ in,
                                                    unsigned short* __restrict__ out, int n) {
    int i = blockIdx.x * 256 + threadIdx.x;
    if (i < n) out[i] = f2bf(in[i]);
}

// ---------------------------------------------------------------------------
// Shared wave-level 16x64 GEMM core with ping-pong K pipeline.
//   acc[c] += A(16xCC) @ B^T where B rows come from `wbase` (weights, row = n).
// arow  : A row pointer for this lane (row mt*16 + l16)
// wbase : weight base pointer already offset by (ntile_row + l16)*CC + 16h
// ---------------------------------------------------------------------------
__device__ __forceinline__ void gemm_core_16x64(const unsigned short* __restrict__ arow,
                                                const unsigned short* __restrict__ wbase,
                                                int h, v8f acc[4]) {
    // ---- prologue: set A <- k = 0
    v16bf aA = load_frag(arow + 16 * h, arow + 8 + 16 * h);
    v16bf bA[4];
#pragma unroll
    for (int c = 0; c < 4; ++c) {
        const unsigned short* wrow = wbase + (size_t)(c * 16) * CC;
        bA[c] = load_frag(wrow, wrow + 8);
    }

    v16bf aB;
    v16bf bB[4];

    // ---- ping-pong main loop: 11 iterations covering k = 0 .. 704
    for (int k0 = 32; k0 + 64 < CC; k0 += 64) {
        // load set B <- k0
        aB = load_frag(arow + k0 + 16 * h, arow + k0 + 8 + 16 * h);
#pragma unroll
        for (int c = 0; c < 4; ++c) {
            const unsigned short* wrow = wbase + (size_t)(c * 16) * CC + k0;
            bB[c] = load_frag(wrow, wrow + 8);
        }
        // consume set A (k0 - 32)
#pragma unroll
        for (int c = 0; c < 4; ++c) acc[c] = WMMA_BF16(aA, bA[c], acc[c]);

        // load set A <- k0 + 32
        aA = load_frag(arow + k0 + 32 + 16 * h, arow + k0 + 40 + 16 * h);
#pragma unroll
        for (int c = 0; c < 4; ++c) {
            const unsigned short* wrow = wbase + (size_t)(c * 16) * CC + k0 + 32;
            bA[c] = load_frag(wrow, wrow + 8);
        }
        // consume set B (k0)
#pragma unroll
        for (int c = 0; c < 4; ++c) acc[c] = WMMA_BF16(aB, bB[c], acc[c]);
    }

    // ---- epilogue: load last step (k = CC-32) into B, consume A then B
    {
        const int kl = CC - 32;
        aB = load_frag(arow + kl + 16 * h, arow + kl + 8 + 16 * h);
#pragma unroll
        for (int c = 0; c < 4; ++c) {
            const unsigned short* wrow = wbase + (size_t)(c * 16) * CC + kl;
            bB[c] = load_frag(wrow, wrow + 8);
        }
#pragma unroll
        for (int c = 0; c < 4; ++c) acc[c] = WMMA_BF16(aA, bA[c], acc[c]);
#pragma unroll
        for (int c = 0; c < 4; ++c) acc[c] = WMMA_BF16(aB, bB[c], acc[c]);
    }
}

// ---------------------------------------------------------------------------
// Stage 1: QKV GEMM.  qkv[m][n] = sum_k x[m][k] * w_attn[n][k] + b_attn[n]
// Epilogue scatters into q[B,H,T,D], k[B,H,T,D], vT[B,H,D,T] (bf16).
// ---------------------------------------------------------------------------
__global__ void __launch_bounds__(128) gemm_qkv(const unsigned short* __restrict__ xb,
                                                const unsigned short* __restrict__ wab,
                                                const float* __restrict__ bias,
                                                unsigned short* __restrict__ qbuf,
                                                unsigned short* __restrict__ kbuf,
                                                unsigned short* __restrict__ vT) {
    const int lane = threadIdx.x & 31;
    const int wave = threadIdx.x >> 5;
    const int gw   = blockIdx.x * 4 + wave;
    const int nt   = gw % (N_QKV / 64);     // 36 N-tiles of 64
    const int mt   = gw / (N_QKV / 64);     // 256 M-tiles of 16
    const int l16  = lane & 15;
    const int h    = lane >> 4;

    const unsigned short* arow  = xb  + (size_t)(mt * 16 + l16) * CC;
    const unsigned short* wbase = wab + (size_t)(nt * 64 + l16) * CC + 16 * h;

    v8f z = {0.f, 0.f, 0.f, 0.f, 0.f, 0.f, 0.f, 0.f};
    v8f acc[4];
#pragma unroll
    for (int c = 0; c < 4; ++c) acc[c] = z;

    gemm_core_16x64(arow, wbase, h, acc);

#pragma unroll
    for (int c = 0; c < 4; ++c) {
        const int n   = nt * 64 + c * 16 + l16;
        const float bv = bias[n];
        const int sec = n / CC;             // 0=q, 1=k, 2=v
        const int rem = n - sec * CC;
        const int hh  = rem >> 6;           // head
        const int d   = rem & 63;
#pragma unroll
        for (int r = 0; r < 8; ++r) {
            const int m  = mt * 16 + r + 8 * h;
            const int bi = m >> 11;         // / TT
            const int t  = m & (TT - 1);
            const unsigned short val = f2bf(acc[c][r] + bv);
            if (sec == 0) {
                qbuf[(((size_t)(bi * HH + hh)) * TT + t) * DD + d] = val;
            } else if (sec == 1) {
                kbuf[(((size_t)(bi * HH + hh)) * TT + t) * DD + d] = val;
            } else {
                vT[(((size_t)(bi * HH + hh)) * DD + d) * TT + t] = val;
            }
        }
    }
}

// ---------------------------------------------------------------------------
// Stage 2: causal flash attention.  One wave owns 16 query rows of one head,
// streams 32 keys per iteration.  K and V fragments are all issued at the top
// of the iteration so the V loads are covered by the softmax VALU work.
// grid = (B*H, T/(16*4)), block = 128 (4 waves).
// ---------------------------------------------------------------------------
__global__ void __launch_bounds__(128) attn_fwd(const unsigned short* __restrict__ qbuf,
                                                const unsigned short* __restrict__ kbuf,
                                                const unsigned short* __restrict__ vT,
                                                unsigned short* __restrict__ ob) {
    __shared__ unsigned short plds[4][16 * 32];   // per-wave P staging (bf16)

    const int lane = threadIdx.x & 31;
    const int w    = threadIdx.x >> 5;
    const int l16  = lane & 15;
    const int h    = lane >> 4;

    const int bh  = blockIdx.x;                 // b*H + head
    const int bi  = bh / HH;
    const int hd  = bh % HH;
    const int qb  = blockIdx.y * 4 + w;         // query block (16 rows)
    const int t0  = qb * 16;

    const unsigned short* qhead = qbuf + (size_t)bh * TT * DD;
    const unsigned short* khead = kbuf + (size_t)bh * TT * DD;
    const unsigned short* vhead = vT   + (size_t)bh * DD * TT;

    // Q fragments (A-matrix), K-dim = d: frag0 d 0..31, frag1 d 32..63
    const unsigned short* qrow = qhead + (size_t)(t0 + l16) * DD;
    const v16bf qf0 = load_frag(qrow +      16 * h, qrow +  8 + 16 * h);
    const v16bf qf1 = load_frag(qrow + 32 + 16 * h, qrow + 40 + 16 * h);

    float mrow[8], lrow[8];
    v8f zero = {0.f, 0.f, 0.f, 0.f, 0.f, 0.f, 0.f, 0.f};
    v8f acc[4];
#pragma unroll
    for (int r = 0; r < 8; ++r) { mrow[r] = -1e30f; lrow[r] = 0.f; }
#pragma unroll
    for (int c = 0; c < 4; ++c) acc[c] = zero;

    for (int s0 = 0; s0 <= t0 + 15; s0 += 32) {
        // ---- issue ALL global loads for this iteration up front:
        //      4 K fragments (needed by S-WMMAs) then 4 V fragments (needed
        //      only after softmax, so their latency hides under VALU work).
        const unsigned short* kr0 = khead + (size_t)(s0 + l16) * DD;
        const unsigned short* kr1 = khead + (size_t)(s0 + 16 + l16) * DD;
        v16bf kb00 = load_frag(kr0 +      16 * h, kr0 +  8 + 16 * h); // d 0..31
        v16bf kb01 = load_frag(kr0 + 32 + 16 * h, kr0 + 40 + 16 * h); // d 32..63
        v16bf kb10 = load_frag(kr1 +      16 * h, kr1 +  8 + 16 * h);
        v16bf kb11 = load_frag(kr1 + 32 + 16 * h, kr1 + 40 + 16 * h);
        v16bf vf[4];
#pragma unroll
        for (int c = 0; c < 4; ++c) {
            const unsigned short* vrow =
                vhead + (size_t)(c * 16 + l16) * TT + s0 + 16 * h;
            vf[c] = load_frag(vrow, vrow + 8);
        }

        // ---- S = q @ k^T (two 16x16 tiles: keys s0..+15 and s0+16..+31)
        v8f S0 = zero, S1 = zero;
        S0 = WMMA_BF16(qf0, kb00, S0);
        S0 = WMMA_BF16(qf1, kb01, S0);
        S1 = WMMA_BF16(qf0, kb10, S1);
        S1 = WMMA_BF16(qf1, kb11, S1);

        // ---- scale + causal mask + online softmax (row r+8h lives in reg r)
        float alpha_r[8];
#pragma unroll
        for (int r = 0; r < 8; ++r) {
            const int t = t0 + r + 8 * h;
            float f0 = S0[r] * ATTN_SCALE;
            float f1 = S1[r] * ATTN_SCALE;
            if (s0 + l16      > t) f0 = -1e30f;
            if (s0 + 16 + l16 > t) f1 = -1e30f;
            float rm = fmaxf(f0, f1);
            rm = fmaxf(rm, __shfl_xor(rm, 1, 32));
            rm = fmaxf(rm, __shfl_xor(rm, 2, 32));
            rm = fmaxf(rm, __shfl_xor(rm, 4, 32));
            rm = fmaxf(rm, __shfl_xor(rm, 8, 32));
            const float m_new = fmaxf(mrow[r], rm);
            const float alpha = __expf(mrow[r] - m_new);
            mrow[r] = m_new;
            const float p0 = __expf(f0 - m_new);
            const float p1 = __expf(f1 - m_new);
            float rs = p0 + p1;
            rs += __shfl_xor(rs, 1, 32);
            rs += __shfl_xor(rs, 2, 32);
            rs += __shfl_xor(rs, 4, 32);
            rs += __shfl_xor(rs, 8, 32);
            lrow[r] = lrow[r] * alpha + rs;
            alpha_r[r] = alpha;
            plds[w][(r + 8 * h) * 32 + l16]      = f2bf(p0);
            plds[w][(r + 8 * h) * 32 + 16 + l16] = f2bf(p1);
        }

        // rescale running output
#pragma unroll
        for (int c = 0; c < 4; ++c)
#pragma unroll
            for (int r = 0; r < 8; ++r) acc[c][r] *= alpha_r[r];

        // wave-private LDS: stores and loads are in-order per wave; make the
        // compiler keep them ordered and drain DScnt (CDNA5 split counters).
        asm volatile("s_wait_dscnt 0" ::: "memory");

        // ---- re-read P as A-fragment, then O += P @ V (V already in regs)
        const unsigned short* prow = &plds[w][l16 * 32];
        const v16bf pA = load_frag(prow + 16 * h, prow + 8 + 16 * h);
#pragma unroll
        for (int c = 0; c < 4; ++c) acc[c] = WMMA_BF16(pA, vf[c], acc[c]);
    }

    // ---- normalize and write [B,T,C] bf16
#pragma unroll
    for (int r = 0; r < 8; ++r) {
        const float inv = 1.0f / lrow[r];
        const int t = t0 + r + 8 * h;
        unsigned short* orow = ob + ((size_t)(bi * TT + t)) * CC + hd * DD;
#pragma unroll
        for (int c = 0; c < 4; ++c) orow[c * 16 + l16] = f2bf(acc[c][r] * inv);
    }
}

// ---------------------------------------------------------------------------
// Stage 3: output projection.  y[m][n] = sum_k o[m][k] * w_proj[n][k] + b[n]
// fp32 output to d_out.
// ---------------------------------------------------------------------------
__global__ void __launch_bounds__(128) gemm_proj(const unsigned short* __restrict__ obuf,
                                                 const unsigned short* __restrict__ wpb,
                                                 const float* __restrict__ bias,
                                                 float* __restrict__ out) {
    const int lane = threadIdx.x & 31;
    const int wave = threadIdx.x >> 5;
    const int gw   = blockIdx.x * 4 + wave;
    const int nt   = gw % (CC / 64);        // 12 N-tiles
    const int mt   = gw / (CC / 64);        // 256 M-tiles
    const int l16  = lane & 15;
    const int h    = lane >> 4;

    const unsigned short* arow  = obuf + (size_t)(mt * 16 + l16) * CC;
    const unsigned short* wbase = wpb  + (size_t)(nt * 64 + l16) * CC + 16 * h;

    v8f z = {0.f, 0.f, 0.f, 0.f, 0.f, 0.f, 0.f, 0.f};
    v8f acc[4];
#pragma unroll
    for (int c = 0; c < 4; ++c) acc[c] = z;

    gemm_core_16x64(arow, wbase, h, acc);

#pragma unroll
    for (int c = 0; c < 4; ++c) {
        const int n = nt * 64 + c * 16 + l16;
        const float bv = bias[n];
#pragma unroll
        for (int r = 0; r < 8; ++r) {
            const int m = mt * 16 + r + 8 * h;
            out[(size_t)m * CC + n] = acc[c][r] + bv;
        }
    }
}

// ---------------------------------------------------------------------------
// Host launcher
// ---------------------------------------------------------------------------
extern "C" void kernel_launch(void* const* d_in, const int* in_sizes, int n_in,
                              void* d_out, int out_size, void* d_ws, size_t ws_size,
                              hipStream_t stream) {
    (void)in_sizes; (void)n_in; (void)out_size; (void)ws_size;

    const float* x      = (const float*)d_in[0];   // [B,T,C]
    const float* w_attn = (const float*)d_in[1];   // [3C,C]
    const float* b_attn = (const float*)d_in[2];   // [3C]
    const float* w_proj = (const float*)d_in[3];   // [C,C]
    const float* b_proj = (const float*)d_in[4];   // [C]
    float* out = (float*)d_out;                    // [B,T,C] fp32

    // Workspace layout (bytes)
    char* ws = (char*)d_ws;
    const size_t SZ_XB  = (size_t)MM * CC * 2;       // 6291456
    const size_t SZ_WAB = (size_t)N_QKV * CC * 2;    // 3538944
    const size_t SZ_WPB = (size_t)CC * CC * 2;       // 1179648
    const size_t SZ_HTD = (size_t)BB * HH * TT * DD * 2; // 6291456

    unsigned short* xb  = (unsigned short*)(ws);
    unsigned short* wab = (unsigned short*)(ws + SZ_XB);
    unsigned short* wpb = (unsigned short*)(ws + SZ_XB + SZ_WAB);
    unsigned short* qb  = (unsigned short*)(ws + SZ_XB + SZ_WAB + SZ_WPB);
    unsigned short* kb  = (unsigned short*)(ws + SZ_XB + SZ_WAB + SZ_WPB + SZ_HTD);
    unsigned short* vT  = (unsigned short*)(ws + SZ_XB + SZ_WAB + SZ_WPB + 2 * SZ_HTD);
    unsigned short* ob  = (unsigned short*)(ws + SZ_XB + SZ_WAB + SZ_WPB + 3 * SZ_HTD);

    // Stage 0: fp32 -> bf16
    {
        int n = MM * CC;
        cvt_f32_bf16<<<(n + 255) / 256, 256, 0, stream>>>(x, xb, n);
        n = N_QKV * CC;
        cvt_f32_bf16<<<(n + 255) / 256, 256, 0, stream>>>(w_attn, wab, n);
        n = CC * CC;
        cvt_f32_bf16<<<(n + 255) / 256, 256, 0, stream>>>(w_proj, wpb, n);
    }

    // Stage 1: QKV GEMM  (256 M-tiles x 36 N-tiles, 4 waves/block)
    gemm_qkv<<<(MM / 16) * (N_QKV / 64) / 4, 128, 0, stream>>>(xb, wab, b_attn, qb, kb, vT);

    // Stage 2: causal attention  (24 heads x 32 groups of 4 query-blocks)
    attn_fwd<<<dim3(BB * HH, TT / 64), 128, 0, stream>>>(qb, kb, vT, ob);

    // Stage 3: projection GEMM  (256 M-tiles x 12 N-tiles, 4 waves/block)
    gemm_proj<<<(MM / 16) * (CC / 64) / 4, 128, 0, stream>>>(ob, wpb, b_proj, out);
}
